// Phase3Stage1Model_4750233829431
// MI455X (gfx1250) — matrix operand order repaired and checked by reading, not compiled, not run on hardware
//
#include <hip/hip_runtime.h>
#include <hip/hip_bf16.h>

// ---------------------------------------------------------------------------
// Complex transformer forward for MI455X (gfx1250, wave32, WMMA f16 -> f32).
// V=32000, D=1024, L=6, S=2048, B=2, N=B*S=4096 tokens.
// Register-blocked WMMA GEMMs: head 32x64/wave, layer 16x32/wave.
// ---------------------------------------------------------------------------

#define VOCAB 32000
#define DIM   1024
#define NLAYER 6
#define SEQ   2048
#define BATCH 2
#define NTOK  (BATCH * SEQ)   // 4096
#define LN_THREADS 256

typedef _Float16 v16h __attribute__((ext_vector_type(16)));
typedef _Float16 v8h  __attribute__((ext_vector_type(8)));
typedef float    v8f  __attribute__((ext_vector_type(8)));

// ---------------- WMMA helper -------------------------------------------------
__device__ __forceinline__ v8f wmma_f16(v16h a, v16h b, v8f c) {
  // D = A(16x32 f16) * B(32x16 f16) + C(16x16 f32)
  return __builtin_amdgcn_wmma_f32_16x16x32_f16(
      /*neg_a=*/false, a, /*neg_b=*/false, b,
      /*c_mod=*/(short)0, c, /*reuse_a=*/false, /*reuse_b=*/false);
}

// A-matrix (16x32, 16-bit) per-lane load. Lane l holds row m0+(l&15).
// Lane-group g = l>>4:  halves 0..7  -> K = kb + g*8 .. +7
//                       halves 8..15 -> K = kb + 16 + g*8 .. +7
// Two contiguous 16-byte chunks -> two global_load_b128.
__device__ __forceinline__ v16h load_a_tile(const _Float16* __restrict__ A,
                                            int row, int ldk, int kb, int g) {
  v16h a;
  const size_t base = (size_t)row * (size_t)ldk;
  ((v8h*)&a)[0] = *(const v8h*)(A + base + kb + g * 8);
  ((v8h*)&a)[1] = *(const v8h*)(A + base + kb + 16 + g * 8);
  return a;
}

// B-matrix (32x16, 16-bit) per-lane load. Lane l holds column e0+(l&15);
// halves 0..7 -> K = kb + g*8 .. +7, halves 8..15 -> K = kb + 16 + g*8 .. +7.
__device__ __forceinline__ v16h load_b_tile(const _Float16* __restrict__ W,
                                            int col, int ldk, int kb, int g) {
  v16h b;
  const _Float16* p = W + (size_t)col * (size_t)ldk + kb + g * 8;
  ((v8h*)&b)[0] = ((const v8h*)p)[0];
  ((v8h*)&b)[1] = ((const v8h*)p)[2];
  return b;
}

// ---------------- elementwise / setup kernels --------------------------------

__global__ void cvt_f32_f16_kernel(const float* __restrict__ src,
                                   _Float16* __restrict__ dst, long n) {
  long i = (long)blockIdx.x * blockDim.x + threadIdx.x;
  long stride = (long)gridDim.x * blockDim.x;
  for (; i < n; i += stride) dst[i] = (_Float16)src[i];
}

__global__ void embed_kernel(const int* __restrict__ ids,
                             const float* __restrict__ emb,     // [2, V, D]
                             const float* __restrict__ pos_emb, // [2, S, D]
                             float* __restrict__ zr, float* __restrict__ zi) {
  long i = (long)blockIdx.x * blockDim.x + threadIdx.x;
  long n = (long)NTOK * DIM;
  long stride = (long)gridDim.x * blockDim.x;
  for (; i < n; i += stride) {
    int t = (int)(i / DIM);
    int d = (int)(i % DIM);
    int tok = ids[t];
    int pos = t % SEQ;
    zr[i] = emb[(size_t)tok * DIM + d] + pos_emb[(size_t)pos * DIM + d];
    zi[i] = emb[(size_t)VOCAB * DIM + (size_t)tok * DIM + d] +
            pos_emb[(size_t)SEQ * DIM + (size_t)pos * DIM + d];
  }
}

// Complex LayerNorm over D. One block (256 threads) per token.
// Writes f32 outputs (nullable) and/or f16 outputs (nullable).
__global__ void complex_ln_kernel(const float* __restrict__ zr,
                                  const float* __restrict__ zi,
                                  const float* __restrict__ p,   // [4, D]
                                  float* outR, float* outI,
                                  _Float16* outRh, _Float16* outIh) {
  const int t = blockIdx.x;
  const float* xr = zr + (size_t)t * DIM;
  const float* xi = zi + (size_t)t * DIM;
  __shared__ float sA[LN_THREADS], sB[LN_THREADS];

  float sr = 0.f, si = 0.f;
  for (int d = threadIdx.x; d < DIM; d += LN_THREADS) { sr += xr[d]; si += xi[d]; }
  sA[threadIdx.x] = sr; sB[threadIdx.x] = si;
  __syncthreads();
  for (int off = LN_THREADS / 2; off > 0; off >>= 1) {
    if (threadIdx.x < off) {
      sA[threadIdx.x] += sA[threadIdx.x + off];
      sB[threadIdx.x] += sB[threadIdx.x + off];
    }
    __syncthreads();
  }
  const float mu_r = sA[0] * (1.0f / DIM);
  const float mu_i = sB[0] * (1.0f / DIM);
  __syncthreads();

  float vv = 0.f;
  for (int d = threadIdx.x; d < DIM; d += LN_THREADS) {
    float cr = xr[d] - mu_r, ci = xi[d] - mu_i;
    vv += cr * cr + ci * ci;
  }
  sA[threadIdx.x] = vv;
  __syncthreads();
  for (int off = LN_THREADS / 2; off > 0; off >>= 1) {
    if (threadIdx.x < off) sA[threadIdx.x] += sA[threadIdx.x + off];
    __syncthreads();
  }
  const float inv = rsqrtf(sA[0] * (1.0f / DIM) + 1e-5f);

  for (int d = threadIdx.x; d < DIM; d += LN_THREADS) {
    float nr = (xr[d] - mu_r) * inv;
    float ni = (xi[d] - mu_i) * inv;
    float gr = p[d], gi = p[DIM + d], br = p[2 * DIM + d], bi = p[3 * DIM + d];
    float orv = nr * gr - ni * gi + br;
    float oiv = nr * gi + ni * gr + bi;
    size_t idx = (size_t)t * DIM + d;
    if (outR)  outR[idx]  = orv;
    if (outI)  outI[idx]  = oiv;
    if (outRh) outRh[idx] = (_Float16)orv;
    if (outIh) outIh[idx] = (_Float16)oiv;
  }
}

// ---------------- complex GEMM + bias + modReLU + residual -------------------
// One wave computes a 16x32 output block (2 complex column tiles).
// o_r = Ar*Wr^T - Ai*Wi^T + br ;  o_i = Ar*Wi^T + Ai*Wr^T + bi
// scale = relu(|o| + mod_b) / (|o| + 1e-6)
// zr/zi (residual in) <- residual + o * scale   (in place; per-element RMW)
__global__ void complex_gemm_modrelu_kernel(
    const _Float16* __restrict__ Ar, const _Float16* __restrict__ Ai,
    const _Float16* __restrict__ Wr, const _Float16* __restrict__ Wi,
    const float* __restrict__ lin_b,   // [2, D]: br | bi
    const float* __restrict__ mod_b,   // [D]
    float* __restrict__ zr, float* __restrict__ zi) {
  const int wid = threadIdx.x >> 5;
  const int lane = threadIdx.x & 31;
  const int colBlocks = DIM >> 5;                      // 32 (blocks of 32 cols)
  const int tile = blockIdx.x * (blockDim.x >> 5) + wid;
  const int total = (NTOK >> 4) * colBlocks;           // 8192
  if (tile >= total) return;                           // wave-uniform guard
  const int rowT = tile / colBlocks, colB = tile % colBlocks;
  const int m0 = rowT << 4, e0 = colB << 5;
  const int g = lane >> 4, ln = lane & 15;
  const int arow = m0 + ln;

  v8f accRR[2] = {{}, {}}, accII[2] = {{}, {}};
  v8f accRI[2] = {{}, {}}, accIR[2] = {{}, {}};
#pragma unroll 2
  for (int kb = 0; kb < DIM; kb += 32) {
    v16h ar = load_a_tile(Ar, arow, DIM, kb, g);
    v16h ai = load_a_tile(Ai, arow, DIM, kb, g);
#pragma unroll
    for (int c = 0; c < 2; ++c) {
      const int bcol = e0 + c * 16 + ln;
      v16h br = load_b_tile(Wr, bcol, DIM, kb, g);
      v16h bi = load_b_tile(Wi, bcol, DIM, kb, g);
      accRR[c] = wmma_f16(ar, br, accRR[c]);
      accII[c] = wmma_f16(ai, bi, accII[c]);
      accRI[c] = wmma_f16(ar, bi, accRI[c]);
      accIR[c] = wmma_f16(ai, br, accIR[c]);
    }
  }

  // C/D layout: lane holds (m = g*8 + v, n = lane&15), v = VGPR index 0..7
#pragma unroll
  for (int c = 0; c < 2; ++c) {
    const int e = e0 + c * 16 + ln;
    const float brb = lin_b[e];
    const float bib = lin_b[DIM + e];
    const float mb  = mod_b[e];
#pragma unroll
    for (int v = 0; v < 8; ++v) {
      const int t = m0 + g * 8 + v;
      float or_ = accRR[c][v] - accII[c][v] + brb;
      float oi_ = accRI[c][v] + accIR[c][v] + bib;
      float mag = sqrtf(or_ * or_ + oi_ * oi_);
      float sc = fmaxf(mag + mb, 0.0f) / (mag + 1e-6f);
      size_t idx = (size_t)t * DIM + e;
      zr[idx] = fmaf(or_, sc, zr[idx]);   // + residual
      zi[idx] = fmaf(oi_, sc, zi[idx]);
    }
  }
}

// ---------------- vocab head GEMM: logits = A * outW^T + out_b ---------------
// One wave computes a 32x64 output block (2 row tiles x 4 col tiles):
// per K-step 12 b128 loads feed 8 WMMAs (1.5 loads/WMMA).
__global__ void head_gemm_kernel(const _Float16* __restrict__ A,   // [N, D] f16
                                 const _Float16* __restrict__ W,   // [V, D] f16
                                 const float* __restrict__ bias,   // [V]
                                 float* __restrict__ out) {        // [N, V]
  const int wid = threadIdx.x >> 5;
  const int lane = threadIdx.x & 31;
  const int colBlocks = VOCAB >> 6;                    // 500 (blocks of 64 cols)
  const long tile = (long)blockIdx.x * (blockDim.x >> 5) + wid;
  const long total = (long)(NTOK >> 5) * colBlocks;    // 64000
  if (tile >= total) return;                           // wave-uniform guard
  const int rowB = (int)(tile / colBlocks), colB = (int)(tile % colBlocks);
  const int m0 = rowB << 5, e0 = colB << 6;
  const int g = lane >> 4, ln = lane & 15;

  v8f acc[2][4] = {{{}, {}, {}, {}}, {{}, {}, {}, {}}};
#pragma unroll 2
  for (int kb = 0; kb < DIM; kb += 32) {
    v16h a0 = load_a_tile(A, m0 + ln,      DIM, kb, g);
    v16h a1 = load_a_tile(A, m0 + 16 + ln, DIM, kb, g);
#pragma unroll
    for (int c = 0; c < 4; ++c) {
      v16h b = load_b_tile(W, e0 + c * 16 + ln, DIM, kb, g);
      acc[0][c] = wmma_f16(a0, b, acc[0][c]);
      acc[1][c] = wmma_f16(a1, b, acc[1][c]);
    }
  }

#pragma unroll
  for (int c = 0; c < 4; ++c) {
    const int e = e0 + c * 16 + ln;
    const float bb = bias[e];
#pragma unroll
    for (int r = 0; r < 2; ++r) {
#pragma unroll
      for (int v = 0; v < 8; ++v) {
        const int t = m0 + r * 16 + g * 8 + v;
        out[(size_t)t * VOCAB + e] = acc[r][c][v] + bb;
      }
    }
  }
}

// ---------------------------------------------------------------------------
extern "C" void kernel_launch(void* const* d_in, const int* in_sizes, int n_in,
                              void* d_out, int out_size, void* d_ws, size_t ws_size,
                              hipStream_t stream) {
  (void)in_sizes; (void)n_in; (void)out_size; (void)ws_size;

  const int*   input_ids = (const int*)  d_in[0];   // [B, S]
  const float* emb       = (const float*)d_in[1];   // [2, V, D]
  const float* pos_emb   = (const float*)d_in[2];   // [2, S, D]
  const float* ln1       = (const float*)d_in[3];   // [L, 4, D]
  const float* lin_W     = (const float*)d_in[4];   // [L, 2, D, D]
  const float* lin_b     = (const float*)d_in[5];   // [L, 2, D]
  const float* mod_b     = (const float*)d_in[6];   // [L, D]
  const float* ln2       = (const float*)d_in[7];   // [L, 4, D]
  const float* fin_ln    = (const float*)d_in[8];   // [4, D]
  const float* out_W     = (const float*)d_in[9];   // [V, D]
  const float* out_b     = (const float*)d_in[10];  // [V]
  float* logits = (float*)d_out;                    // [B, S, V]

  // ---- workspace carve (all chunks 16B-aligned; ~120 MB total) ----
  const size_t nd = (size_t)NTOK * DIM;
  char* w = (char*)d_ws;
  float*    zr  = (float*)w;    w += nd * sizeof(float);
  float*    zi  = (float*)w;    w += nd * sizeof(float);
  _Float16* zrh = (_Float16*)w; w += nd * sizeof(_Float16);
  _Float16* zih = (_Float16*)w; w += nd * sizeof(_Float16);
  _Float16* Whr = (_Float16*)w; w += (size_t)DIM * DIM * sizeof(_Float16);
  _Float16* Whi = (_Float16*)w; w += (size_t)DIM * DIM * sizeof(_Float16);
  _Float16* oWh = (_Float16*)w; w += (size_t)VOCAB * DIM * sizeof(_Float16);

  // 1) out_W -> f16 (65 MB, fits in the 192 MB L2 for the head GEMM)
  cvt_f32_f16_kernel<<<2048, 256, 0, stream>>>(out_W, oWh, (long)VOCAB * DIM);

  // 2) embeddings
  embed_kernel<<<4096, 256, 0, stream>>>(input_ids, emb, pos_emb, zr, zi);

  const int gemmBlocks = ((NTOK / 16) * (DIM / 32)) / 8;   // 1024 (8 waves/block)

  // 3) layers
  for (int l = 0; l < NLAYER; ++l) {
    const float* Wl  = lin_W + (size_t)l * 2 * DIM * DIM;
    cvt_f32_f16_kernel<<<1024, 256, 0, stream>>>(Wl,                      Whr, (long)DIM * DIM);
    cvt_f32_f16_kernel<<<1024, 256, 0, stream>>>(Wl + (size_t)DIM * DIM, Whi, (long)DIM * DIM);

    // LN1: only f16 outputs needed (feeds WMMA); zr/zi stay = residual
    complex_ln_kernel<<<NTOK, LN_THREADS, 0, stream>>>(
        zr, zi, ln1 + (size_t)l * 4 * DIM, nullptr, nullptr, zrh, zih);

    // complex linear + bias + modReLU + residual (in-place into zr/zi)
    complex_gemm_modrelu_kernel<<<gemmBlocks, 256, 0, stream>>>(
        zrh, zih, Whr, Whi,
        lin_b + (size_t)l * 2 * DIM, mod_b + (size_t)l * DIM, zr, zi);

    // LN2: f32 in-place
    complex_ln_kernel<<<NTOK, LN_THREADS, 0, stream>>>(
        zr, zi, ln2 + (size_t)l * 4 * DIM, zr, zi, nullptr, nullptr);
  }

  // 4) final LN -> f16 real part only (head uses zr only)
  complex_ln_kernel<<<NTOK, LN_THREADS, 0, stream>>>(
      zr, zi, fin_ln, nullptr, nullptr, zrh, nullptr);

  // 5) vocab head: 64000 wave-tiles of 32x64
  const int headBlocks = ((NTOK / 32) * (VOCAB / 64)) / 8;  // 8000
  head_gemm_kernel<<<headBlocks, 256, 0, stream>>>(zrh, oWh, out_b, logits);
}
